// PolicyEstimator_23356032155839
// MI455X (gfx1250) — matrix operand order repaired and checked by reading, not compiled, not run on hardware
//
#include <hip/hip_runtime.h>
#include <hip/hip_bf16.h>
#include <math.h>

// ---------------------------------------------------------------------------
// Autoregressive 2-layer LSTM policy head on MI455X (gfx1250).
//
// Strategy (bandwidth-roofline driven):
//  * GEMV-dominated, AI ~0.5 FLOP/B -> memory bound. fp32 weights = 272 MiB/step.
//  * Convert all weights to bf16 ONCE per call (deterministic) into d_ws:
//    136 MiB working set fits the 192 MB global L2, so all 16 steps re-read
//    from L2 instead of HBM. Halves traffic AND enables v_wmma bf16 path.
//    The one-shot fp32 source reads use NON-TEMPORAL loads so they do not
//    evict the freshly written bf16 pack from L2.
//  * Weights are pre-packed into WMMA A-fragments in this code's contiguous-K order
//    (16x32 bf16 tiles; lanes 0-15 hold row M=lane,
//    K [0..15]; lanes 16-31 hold K [16..31]),
//    so the hot loop is: 2x global_load_b128 (A) + 2x ds_load_b128 (B)
//    + v_wmma_f32_16x16x32_bf16, fully coalesced (verified in disasm:
//    s_clause'd b128 loads, partial s_wait counts, 8 wmma per iteration).
//  * The input vector chunk is replicated into all 16 B columns, so
//    D(M,N)=y[M] for every N; lane 0 holds y[0..7] in acc v0..7,
//    lane 16 holds y[8..15] (32-bit C/D layout, ISA 7.12.2).
//  * LSTM cell fused into the GEMV kernel: one block owns 16 hidden units
//    and all 4 gates (i,f,g,o row blocks of W), 16 waves = 4 gates x 4
//    K-quarters over concatenated K=4096 ([W_ih | W_hh] rows, [x ; h] vector).
//  * Sequencing via stream order: 4 kernels/step x 16 steps (+4 prep).
//    h double-buffered across steps (cross-block WAR); c in-place (owner-only).
//  * No atomics, fixed reduction order -> bit-deterministic across replays.
//
// Workspace budget (assumed available): 2x64 MiB layer packs + 8 MiB W_out
// pack + ~128 KiB state ~= 136.2 MiB.
// ---------------------------------------------------------------------------

typedef __attribute__((ext_vector_type(16))) __bf16 v16bf;
typedef __attribute__((ext_vector_type(8)))  float  v8f;

union AFrag { uint4 q[2]; v16bf v; };

#define HID   2048
#define STEPS 16

__device__ __forceinline__ float sig_(float x) { return 1.0f / (1.0f + expf(-x)); }

// ---------------------------------------------------------------------------
// Weight packing: dst tile t = (g*128 + rt)*128 + kt, 512 bf16 elems per tile.
// Element e in tile: lane = e>>4, j = e&15, seg = j>>3, pos = j&7.
// lane<16: M=lane,  K = (seg?8:0)+pos ; lane>=16: M=lane-16, K = (seg?24:16)+pos.
// Concatenated K: kt<64 -> W_ih col (kt*32+K), else W_hh col (kt*32+K-2048).
// fp32 sources are read non-temporally (read-once) to protect the bf16 pack
// residency in L2.
// ---------------------------------------------------------------------------
__global__ __launch_bounds__(256) void pack_layer_k(
    const float* __restrict__ Wih, const float* __restrict__ Whh,
    __bf16* __restrict__ dst)
{
    const size_t NL = (size_t)4 * 128 * 128 * 512;   // 33.55M bf16 per layer
    for (size_t id = (size_t)blockIdx.x * blockDim.x + threadIdx.x; id < NL;
         id += (size_t)gridDim.x * blockDim.x) {
        const int    e    = (int)(id & 511);
        const size_t tile = id >> 9;
        const int    kt   = (int)(tile & 127);
        const int    grt  = (int)(tile >> 7);          // g*128 + rt
        const int lane = e >> 4, jj = e & 15, seg = jj >> 3, pos = jj & 7;
        const int M     = lane & 15;
        const int kbase = (lane < 16) ? (seg ? 8 : 0) : (seg ? 24 : 16);
        const int row   = (grt >> 7) * 2048 + (grt & 127) * 16 + M; // 0..8191
        const int kk    = kt * 32 + kbase + pos;                    // 0..4095
        const float v = (kk < 2048)
            ? __builtin_nontemporal_load(Wih + (size_t)row * 2048 + kk)
            : __builtin_nontemporal_load(Whh + (size_t)row * 2048 + (kk - 2048));
        dst[id] = (__bf16)v;
    }
}

__global__ __launch_bounds__(256) void pack_out_k(
    const float* __restrict__ W, __bf16* __restrict__ dst)
{
    const size_t NO = (size_t)128 * 64 * 512;        // 4.19M bf16
    for (size_t id = (size_t)blockIdx.x * blockDim.x + threadIdx.x; id < NO;
         id += (size_t)gridDim.x * blockDim.x) {
        const int    e    = (int)(id & 511);
        const size_t tile = id >> 9;
        const int    kt   = (int)(tile & 63);
        const int    rt   = (int)(tile >> 6);
        const int lane = e >> 4, jj = e & 15, seg = jj >> 3, pos = jj & 7;
        const int M     = lane & 15;
        const int kbase = (lane < 16) ? (seg ? 8 : 0) : (seg ? 24 : 16);
        const int row = rt * 16 + M;
        const int kk  = kt * 32 + kbase + pos;
        dst[id] = (__bf16)__builtin_nontemporal_load(W + (size_t)row * 2048 + kk);
    }
}

// bsum = b_ih + b_hh (both layers), zero h buffers + c, x -> bf16.
__global__ __launch_bounds__(256) void init_k(
    const float* __restrict__ x, const float* __restrict__ b_ih,
    const float* __restrict__ b_hh, float* __restrict__ bsum,
    __bf16* __restrict__ hbuf, float* __restrict__ cbuf,
    __bf16* __restrict__ xvec)
{
    const int i = blockIdx.x * blockDim.x + threadIdx.x;
    if (i < 16384)                       bsum[i] = b_ih[i] + b_hh[i];
    else if (i < 16384 + 8192)           hbuf[i - 16384] = (__bf16)0.0f;
    else if (i < 16384 + 8192 + 4096)    cbuf[i - 16384 - 8192] = 0.0f;
    else if (i < 16384 + 8192 + 4096 + 2048)
        xvec[i - 16384 - 8192 - 4096] = (__bf16)x[i - 16384 - 8192 - 4096];
}

// ---------------------------------------------------------------------------
// One LSTM layer for one timestep. Block rt owns hidden units [rt*16, rt*16+16).
// 16 waves: wave wv -> gate g = wv&3, K-quarter kq = wv>>2 (32 ktiles each,
// K total = 4096 = [x(2048) ; h(2048)]). Fused cell update at the end.
// ---------------------------------------------------------------------------
__global__ __launch_bounds__(512) void lstm_layer_k(
    const __bf16* __restrict__ packW,   // this layer's packed [Wih|Whh]
    const float*  __restrict__ bsum,    // this layer's b_ih+b_hh (8192)
    const __bf16* __restrict__ xin,     // layer input vec (2048 bf16)
    const __bf16* __restrict__ hin,     // h_{t-1}         (2048 bf16)
    float*        __restrict__ cio,     // c (fp32, in-place, owner-only)
    __bf16*       __restrict__ hout)    // h_t             (2048 bf16)
{
    __shared__ __align__(64) __bf16 xh[4096];
    __shared__ float part[16][16];

    const int tid = threadIdx.x;
    {   // cooperative vector copy of [x ; h] into LDS (512 x uint4 = 8 KiB)
        uint4* d = (uint4*)xh;
        if (tid < 256) d[tid] = ((const uint4*)xin)[tid];
        else           d[tid] = ((const uint4*)hin)[tid - 256];
    }
    __syncthreads();

    const int wv   = tid >> 5;
    const int lane = tid & 31;
    const int g    = wv & 3;
    const int kq   = wv >> 2;
    const int rt   = blockIdx.x;     // 0..127

    // A: per-lane 32B of packed fragment; B: replicated x/h chunk from LDS.
    const __bf16* ap = packW + (((size_t)(g * 128 + rt)) * 128 + kq * 32) * 512
                             + lane * 16;
    const __bf16* bp = xh + kq * 32 * 32 + ((lane < 16) ? 0 : 16);

    v8f acc = {0.f, 0.f, 0.f, 0.f, 0.f, 0.f, 0.f, 0.f};
    #pragma unroll 8
    for (int i = 0; i < 32; ++i) {
        AFrag a;
        const uint4* aq = (const uint4*)(ap + (size_t)i * 512);
        a.q[0] = aq[0];
        a.q[1] = aq[1];
        const v16bf b = *(const v16bf*)(bp + i * 32);
        acc = __builtin_amdgcn_wmma_f32_16x16x32_bf16(
            false, a.v, false, b, (short)0, acc, false, false);
    }

    // D columns are identical (B replicated): lane0 v0..7 = y[0..7],
    // lane16 v0..7 = y[8..15].
    if (lane == 0) {
        #pragma unroll
        for (int r = 0; r < 8; ++r) part[wv][r] = acc[r];
    } else if (lane == 16) {
        #pragma unroll
        for (int r = 0; r < 8; ++r) part[wv][8 + r] = acc[r];
    }
    __syncthreads();

    if (tid < 16) {                       // fused LSTM cell (PyTorch i,f,g,o)
        const int j = rt * 16 + tid;
        const float si = part[0][tid] + part[4][tid] + part[ 8][tid] + part[12][tid] + bsum[0 * HID + j];
        const float sf = part[1][tid] + part[5][tid] + part[ 9][tid] + part[13][tid] + bsum[1 * HID + j];
        const float sg = part[2][tid] + part[6][tid] + part[10][tid] + part[14][tid] + bsum[2 * HID + j];
        const float so = part[3][tid] + part[7][tid] + part[11][tid] + part[15][tid] + bsum[3 * HID + j];
        const float c = sig_(sf) * cio[j] + sig_(si) * tanhf(sg);
        cio[j]  = c;
        hout[j] = (__bf16)(sig_(so) * tanhf(c));
    }
}

// logits = W_out @ top + b_out. Block rt owns rows [rt*16, rt*16+16);
// 8 waves = 8 K-splits of 8 ktiles each (K = 2048).
__global__ __launch_bounds__(256) void out_gemv_k(
    const __bf16* __restrict__ packO, const __bf16* __restrict__ top,
    const float* __restrict__ bout, float* __restrict__ logits)
{
    __shared__ __align__(64) __bf16 xv[2048];
    __shared__ float part[8][16];
    const int tid = threadIdx.x;
    ((uint4*)xv)[tid] = ((const uint4*)top)[tid];   // 256 x uint4 = 4 KiB
    __syncthreads();

    const int wv   = tid >> 5;
    const int lane = tid & 31;
    const int rt   = blockIdx.x;

    const __bf16* ap = packO + (((size_t)rt) * 64 + wv * 8) * 512 + lane * 16;
    const __bf16* bp = xv + wv * 8 * 32 + ((lane < 16) ? 0 : 16);

    v8f acc = {0.f, 0.f, 0.f, 0.f, 0.f, 0.f, 0.f, 0.f};
    #pragma unroll
    for (int i = 0; i < 8; ++i) {
        AFrag a;
        const uint4* aq = (const uint4*)(ap + (size_t)i * 512);
        a.q[0] = aq[0];
        a.q[1] = aq[1];
        const v16bf b = *(const v16bf*)(bp + i * 32);
        acc = __builtin_amdgcn_wmma_f32_16x16x32_bf16(
            false, a.v, false, b, (short)0, acc, false, false);
    }
    if (lane == 0) {
        #pragma unroll
        for (int r = 0; r < 8; ++r) part[wv][r] = acc[r];
    } else if (lane == 16) {
        #pragma unroll
        for (int r = 0; r < 8; ++r) part[wv][8 + r] = acc[r];
    }
    __syncthreads();

    if (tid < 16) {
        const int j = rt * 16 + tid;
        float s = bout[j];
        #pragma unroll
        for (int w = 0; w < 8; ++w) s += part[w][tid];
        logits[j] = s;
    }
}

// Single-block softmax over 2048 logits; writes fp32 preds to d_out and
// bf16 preds back as next step's input vector. Fixed-order reductions.
// d_out is never re-read on device -> non-temporal stores.
__global__ __launch_bounds__(256) void softmax_k(
    const float* __restrict__ logits, float* __restrict__ outp,
    __bf16* __restrict__ xnext)
{
    __shared__ float red[256];
    __shared__ float ex[2048];
    const int t = threadIdx.x;

    float m = -3.402823466e38f;
    for (int i = t; i < 2048; i += 256) m = fmaxf(m, logits[i]);
    red[t] = m; __syncthreads();
    for (int s = 128; s > 0; s >>= 1) {
        if (t < s) red[t] = fmaxf(red[t], red[t + s]);
        __syncthreads();
    }
    const float gmax = red[0];
    __syncthreads();

    float acc = 0.f;
    for (int i = t; i < 2048; i += 256) {
        const float e = expf(logits[i] - gmax);
        ex[i] = e;
        acc += e;
    }
    red[t] = acc; __syncthreads();
    for (int s = 128; s > 0; s >>= 1) {
        if (t < s) red[t] += red[t + s];
        __syncthreads();
    }
    const float inv = 1.0f / red[0];
    for (int i = t; i < 2048; i += 256) {
        const float p = ex[i] * inv;
        __builtin_nontemporal_store(p, outp + i);
        xnext[i] = (__bf16)p;
    }
}

// ---------------------------------------------------------------------------
extern "C" void kernel_launch(void* const* d_in, const int* in_sizes, int n_in,
                              void* d_out, int out_size, void* d_ws, size_t ws_size,
                              hipStream_t stream)
{
    (void)in_sizes; (void)n_in; (void)out_size; (void)ws_size;

    const float* x    = (const float*)d_in[0];          // [2048]
    const float* Wih  = (const float*)d_in[1];          // [2,8192,2048]
    const float* Whh  = (const float*)d_in[2];          // [2,8192,2048]
    const float* bih  = (const float*)d_in[3];          // [2,8192]
    const float* bhh  = (const float*)d_in[4];          // [2,8192]
    const float* Wout = (const float*)d_in[5];          // [2048,2048]
    const float* bout = (const float*)d_in[6];          // [2048]
    float* out = (float*)d_out;                         // [16,2048]

    // Workspace carve (all offsets 16B+ aligned; total ~136.2 MiB).
    char* p = (char*)d_ws;
    const size_t NLb = (size_t)4 * 128 * 128 * 512 * 2;     // 64 MiB / layer
    __bf16* packL0 = (__bf16*)p; p += NLb;
    __bf16* packL1 = (__bf16*)p; p += NLb;
    __bf16* packO  = (__bf16*)p; p += (size_t)128 * 64 * 512 * 2; // 8 MiB
    float*  bsum   = (float*)p;  p += (size_t)16384 * 4;
    __bf16* hbuf   = (__bf16*)p; p += (size_t)4 * 2048 * 2;  // 2 parities x 2 layers
    float*  cbuf   = (float*)p;  p += (size_t)2 * 2048 * 4;
    __bf16* xvec   = (__bf16*)p; p += (size_t)2048 * 2;
    float*  logits = (float*)p;  p += (size_t)2048 * 4;

    // Per-call prep (deterministic; reads inputs fresh every call).
    pack_layer_k<<<8192, 256, 0, stream>>>(Wih, Whh, packL0);
    pack_layer_k<<<8192, 256, 0, stream>>>(Wih + (size_t)8192 * 2048,
                                           Whh + (size_t)8192 * 2048, packL1);
    pack_out_k<<<2048, 256, 0, stream>>>(Wout, packO);
    init_k<<<(16384 + 8192 + 4096 + 2048 + 255) / 256, 256, 0, stream>>>(
        x, bih, bhh, bsum, hbuf, cbuf, xvec);

    for (int step = 0; step < STEPS; ++step) {
        const int par = step & 1;
        __bf16* hp0 = hbuf + ((size_t)(par * 2 + 0)) * 2048;   // h prev, layer0
        __bf16* hp1 = hbuf + ((size_t)(par * 2 + 1)) * 2048;   // h prev, layer1
        __bf16* hn0 = hbuf + ((size_t)((par ^ 1) * 2 + 0)) * 2048; // h new
        __bf16* hn1 = hbuf + ((size_t)((par ^ 1) * 2 + 1)) * 2048;

        lstm_layer_k<<<128, 512, 0, stream>>>(packL0, bsum,        xvec, hp0,
                                              cbuf,        hn0);
        lstm_layer_k<<<128, 512, 0, stream>>>(packL1, bsum + 8192, hn0,  hp1,
                                              cbuf + 2048, hn1);
        out_gemv_k<<<128, 256, 0, stream>>>(packO, hn1, bout, logits);
        softmax_k<<<1, 256, 0, stream>>>(logits, out + (size_t)step * 2048, xvec);
    }
}
